// VQModel_54176717472043
// MI455X (gfx1250) — compile-verified, matrix operand order checked by reading, not run on hardware
//
#include <hip/hip_runtime.h>
#include <math.h>

// ---------- types ----------
typedef __attribute__((ext_vector_type(16))) __bf16 v16bf;
typedef __attribute__((ext_vector_type(8)))  float  v8f;

union Frag {
    v16bf v;
    uint4 u[2];
};

#define BATCH 131072
#define HID   512
#define EMBD  128
#define NCODE 1024
#define OUTD  6

// ---------- bf16 helpers (round-to-nearest-even) ----------
static __device__ __forceinline__ unsigned short f2bf(float f) {
    unsigned u = __float_as_uint(f);
    unsigned r = u + 0x7FFFu + ((u >> 16) & 1u);
    return (unsigned short)(r >> 16);
}
static __device__ __forceinline__ float bf2f(unsigned short h) {
    return __uint_as_float(((unsigned)h) << 16);
}

// ---------- weight transpose + bf16 convert:  Wt[n][k] = bf16(W[k][n]) ----------
__global__ void wtrans_kernel(const float* __restrict__ W, unsigned short* __restrict__ Wt,
                              int K, int N) {
    int id = blockIdx.x * 256 + threadIdx.x;
    if (id >= K * N) return;
    int n = id / K, k = id % K;               // coalesced writes along k
    Wt[n * K + k] = f2bf(W[k * N + n]);
}

// ---------- codebook prep: bf16 copy + e2[k] = sum_d emb[k,d]^2 ----------
__global__ void embprep_kernel(const float* __restrict__ emb, unsigned short* __restrict__ embbf,
                               float* __restrict__ e2) {
    __shared__ float red[128];
    int k = blockIdx.x, d = threadIdx.x;      // 1024 blocks x 128 threads
    float v = emb[k * EMBD + d];
    embbf[k * EMBD + d] = f2bf(v);
    red[d] = v * v;
    __syncthreads();
    for (int off = 64; off > 0; off >>= 1) {
        if (d < off) red[d] += red[d + off];
        __syncthreads();
    }
    if (d == 0) e2[k] = red[0];
}

// ---------- zero histogram + loss accumulator ----------
__global__ void initz_kernel(float* __restrict__ counts, float* __restrict__ lossAcc) {
    int i = blockIdx.x * 256 + threadIdx.x;
    if (i < NCODE) counts[i] = 0.0f;
    if (i == 0) *lossAcc = 0.0f;
}

// ---------- encoder layer 1: [B,6] x [6,512] + b, relu -> bf16 ----------
__global__ __launch_bounds__(256) void enc1_kernel(const float* __restrict__ x,
                                                   const float* __restrict__ W4,
                                                   const float* __restrict__ b4,
                                                   unsigned short* __restrict__ h1) {
    int b = blockIdx.x >> 1;
    int c = ((blockIdx.x & 1) << 8) + threadIdx.x;
    const float* xr = x + b * 6;              // [B,2,3] row-major == 6 contiguous floats
    float acc = b4[c];
#pragma unroll
    for (int k = 0; k < 6; ++k) acc += xr[k] * W4[k * HID + c];
    h1[b * HID + c] = f2bf(fmaxf(acc, 0.0f));
}

// ---------- generic bf16 WMMA GEMM:  O[B,N] = relu?(A[B,K] @ W[K,N] + bias) ----------
// W pre-transposed as Wt[N][K] bf16. Block: 256 thr = 8 waves, 32-row M tile.
// Double-buffered LDS A-tile (1 barrier per K-step) + double-buffered B fragments
// (static indices only — fully unrolled nt loop).
template <int KDIM, int N, bool RELU, bool ALSO_F32>
__global__ __launch_bounds__(256) void gemm_bf16_kernel(const unsigned short* __restrict__ A,
                                                        const unsigned short* __restrict__ Wt,
                                                        const float* __restrict__ bias,
                                                        unsigned short* __restrict__ Obf,
                                                        float* __restrict__ Of32) {
    constexpr int NPER = N / 8;
    constexpr int NT   = NPER / 16;
    __shared__ __align__(16) unsigned short ldsA[2][32 * 40];  // padded stride 40 (80B rows)

    const int tid  = threadIdx.x;
    const int wave = tid >> 5, lane = tid & 31;
    const int m0    = blockIdx.x * 32;
    const int nbase = wave * NPER;
    const int lcol  = lane & 15;
    const int koff  = (lane < 16) ? 0 : 8;    // ISA bf16 A/B fragment K-split

    const int arow = tid >> 3, akq = (tid & 7) << 2;           // staging decomposition
    const unsigned short* aSrc = &A[(size_t)(m0 + arow) * KDIM + akq];

    auto stageA = [&](int buf, int ks) {
        *(uint2*)&ldsA[buf][arow * 40 + akq] = *(const uint2*)(aSrc + ks);
    };
    auto loadB = [&](Frag& f, int nt, int ks) {
        const unsigned short* p = &Wt[(size_t)(nbase + nt * 16 + lcol) * KDIM + ks + koff];
        f.u[0] = *(const uint4*)p;
        f.u[1] = *(const uint4*)(p + 16);
    };

    v8f acc[2][NT];
#pragma unroll
    for (int ms = 0; ms < 2; ++ms)
#pragma unroll
        for (int nt = 0; nt < NT; ++nt)
#pragma unroll
            for (int j = 0; j < 8; ++j) acc[ms][nt][j] = 0.0f;

    stageA(0, 0);   // preload first A tile

    for (int ks = 0; ks < KDIM; ks += 32) {
        const int cur = (ks >> 5) & 1;
        __syncthreads();                       // staging of `cur` complete; prior reads done
        if (ks + 32 < KDIM) {
            stageA(cur ^ 1, ks + 32);          // fill other buffer while reading this one
            if (ks + 64 < KDIM)
                __builtin_prefetch(aSrc + ks + 64, 0, 0);
        }

        Frag afr[2];
#pragma unroll
        for (int ms = 0; ms < 2; ++ms) {
            const unsigned short* p = &ldsA[cur][(ms * 16 + lcol) * 40 + koff];
            afr[ms].u[0] = *(const uint4*)p;          // K = koff .. koff+7
            afr[ms].u[1] = *(const uint4*)(p + 16);   // K = koff+16 .. koff+23
        }

        Frag bfr[2];
        loadB(bfr[0], 0, ks);
#pragma unroll
        for (int nt = 0; nt < NT; ++nt) {       // fully unrolled -> static bfr indices
            if (nt + 1 < NT) loadB(bfr[(nt + 1) & 1], nt + 1, ks);
#pragma unroll
            for (int ms = 0; ms < 2; ++ms)
                acc[ms][nt] = __builtin_amdgcn_wmma_f32_16x16x32_bf16(
                    false, afr[ms].v, false, bfr[nt & 1].v, (short)0, acc[ms][nt], false, false);
        }
    }

    // epilogue: C layout — VGPR v, lanes 0-15: row v ; lanes 16-31: row v+8
#pragma unroll
    for (int ms = 0; ms < 2; ++ms)
#pragma unroll
        for (int nt = 0; nt < NT; ++nt) {
            const int col = nbase + nt * 16 + lcol;
            const float bv = bias[col];
#pragma unroll
            for (int v = 0; v < 8; ++v) {
                int row = m0 + ms * 16 + v + ((lane < 16) ? 0 : 8);
                float val = acc[ms][nt][v] + bv;
                if (RELU) val = fmaxf(val, 0.0f);
                Obf[(size_t)row * N + col] = f2bf(val);
                if (ALSO_F32) Of32[(size_t)row * N + col] = val;
            }
        }
}

// ---------- VQ scores + argmin:  argmin_k ( e2[k] - 2 * z . emb[k] ) ----------
// Double-buffered codebook fragments with NAMED buffers + 2x-unrolled loop so all
// private-array indices are compile-time constants (no scratch demotion).
__global__ __launch_bounds__(256) void vq_argmin_kernel(const unsigned short* __restrict__ zbf,
                                                        const unsigned short* __restrict__ embbf,
                                                        const float* __restrict__ e2,
                                                        int* __restrict__ idxOut) {
    const int tid = threadIdx.x, wave = tid >> 5, lane = tid & 31;
    const int r0   = (blockIdx.x * 8 + wave) * 16;
    const int lcol = lane & 15;
    const int koff = (lane < 16) ? 0 : 8;

    // z A-fragments for all of D=128 (4 K-steps) held in registers
    Frag afr[4];
    const unsigned short* zp = zbf + (size_t)(r0 + lcol) * EMBD + koff;
#pragma unroll
    for (int kk = 0; kk < 4; ++kk) {
        afr[kk].u[0] = *(const uint4*)(zp + kk * 32);
        afr[kk].u[1] = *(const uint4*)(zp + kk * 32 + 16);
    }

    auto loadBTile = [&](Frag* f, int ct) {
        const unsigned short* ep = embbf + (size_t)(ct * 16 + lcol) * EMBD + koff;
#pragma unroll
        for (int kk = 0; kk < 4; ++kk) {
            f[kk].u[0] = *(const uint4*)(ep + kk * 32);
            f[kk].u[1] = *(const uint4*)(ep + kk * 32 + 16);
        }
    };

    float best[8];
    int   bidx[8];
#pragma unroll
    for (int v = 0; v < 8; ++v) { best[v] = 3.4e38f; bidx[v] = 0; }

    auto scoreTile = [&](const Frag* bf, int ct) {
        v8f acc;
#pragma unroll
        for (int j = 0; j < 8; ++j) acc[j] = 0.0f;
#pragma unroll
        for (int kk = 0; kk < 4; ++kk)
            acc = __builtin_amdgcn_wmma_f32_16x16x32_bf16(
                false, afr[kk].v, false, bf[kk].v, (short)0, acc, false, false);
        const int code = ct * 16 + lcol;
        const float ecol = e2[code];
#pragma unroll
        for (int v = 0; v < 8; ++v) {
            float s = ecol - 2.0f * acc[v];
            if (s < best[v]) { best[v] = s; bidx[v] = code; }
        }
    };

    Frag bA[4], bB[4];                 // two named buffers, only static indexing
    loadBTile(bA, 0);

    for (int ct = 0; ct < NCODE / 16; ct += 2) {
        loadBTile(bB, ct + 1);         // tile ct+1 loads in flight during tile ct WMMAs
        scoreTile(bA, ct);
        if (ct + 2 < NCODE / 16) loadBTile(bA, ct + 2);
        scoreTile(bB, ct + 1);
    }

    // reduce across the 16 lanes of each half (same row within each half)
#pragma unroll
    for (int m = 1; m < 16; m <<= 1) {
#pragma unroll
        for (int v = 0; v < 8; ++v) {
            float ob = __shfl_xor(best[v], m, 32);
            int   oi = __shfl_xor(bidx[v], m, 32);
            if (ob < best[v] || (ob == best[v] && oi < bidx[v])) { best[v] = ob; bidx[v] = oi; }
        }
    }
    if (lcol == 0) {
        int rbase = r0 + ((lane < 16) ? 0 : 8);
#pragma unroll
        for (int v = 0; v < 8; ++v) idxOut[rbase + v] = bidx[v];
    }
}

// ---------- gather q (bf16), loss partial sum, histogram ----------
__global__ __launch_bounds__(256) void vq_gather_kernel(const int* __restrict__ idx,
                                                        const float* __restrict__ emb,
                                                        const float* __restrict__ zf,
                                                        unsigned short* __restrict__ qbf,
                                                        float* __restrict__ counts,
                                                        float* __restrict__ lossAcc) {
    __shared__ float red[256];
    int tid = threadIdx.x;
    int row = blockIdx.x * 256 + tid;
    int k = idx[row];
    const float4* ep = (const float4*)(emb + (size_t)k * EMBD);
    const float4* zp = (const float4*)(zf + (size_t)row * EMBD);
    unsigned short* qp = qbf + (size_t)row * EMBD;
    float s = 0.0f;
#pragma unroll 4
    for (int i = 0; i < EMBD / 4; ++i) {
        float4 e = ep[i], z = zp[i];
        float d0 = e.x - z.x, d1 = e.y - z.y, d2 = e.z - z.z, d3 = e.w - z.w;
        s += d0 * d0 + d1 * d1 + d2 * d2 + d3 * d3;
        qp[i * 4 + 0] = f2bf(e.x);
        qp[i * 4 + 1] = f2bf(e.y);
        qp[i * 4 + 2] = f2bf(e.z);
        qp[i * 4 + 3] = f2bf(e.w);
    }
    atomicAdd(&counts[k], 1.0f);
    red[tid] = s;
    __syncthreads();
    for (int off = 128; off > 0; off >>= 1) {
        if (tid < off) red[tid] += red[tid + off];
        __syncthreads();
    }
    if (tid == 0) atomicAdd(lossAcc, red[0]);
}

// ---------- decoder output layer: [B,512] bf16 x [512,6] + b3 -> f32 ----------
__global__ __launch_bounds__(256) void dec_out_kernel(const unsigned short* __restrict__ g2,
                                                      const float* __restrict__ W3,
                                                      const float* __restrict__ b3,
                                                      float* __restrict__ out) {
    __shared__ float w[HID * OUTD];
    for (int i = threadIdx.x; i < HID * OUTD; i += 256) w[i] = W3[i];
    __syncthreads();
    int row = blockIdx.x * 256 + threadIdx.x;
    const unsigned short* gp = g2 + (size_t)row * HID;
    float a0 = 0, a1 = 0, a2 = 0, a3 = 0, a4 = 0, a5 = 0;
    for (int kb = 0; kb < HID / 8; ++kb) {
        uint4 u = *(const uint4*)(gp + kb * 8);
        unsigned pk[4] = {u.x, u.y, u.z, u.w};
#pragma unroll
        for (int j = 0; j < 4; ++j) {
            float h0 = bf2f((unsigned short)(pk[j] & 0xFFFFu));
            float h1 = bf2f((unsigned short)(pk[j] >> 16));
            const float* w0 = &w[(kb * 8 + j * 2) * OUTD];
            a0 += h0 * w0[0]; a1 += h0 * w0[1]; a2 += h0 * w0[2];
            a3 += h0 * w0[3]; a4 += h0 * w0[4]; a5 += h0 * w0[5];
            const float* w1 = w0 + OUTD;
            a0 += h1 * w1[0]; a1 += h1 * w1[1]; a2 += h1 * w1[2];
            a3 += h1 * w1[3]; a4 += h1 * w1[4]; a5 += h1 * w1[5];
        }
    }
    float* o = out + 1 + (size_t)row * OUTD;   // out[0] is loss
    o[0] = a0 + b3[0]; o[1] = a1 + b3[1]; o[2] = a2 + b3[2];
    o[3] = a3 + b3[3]; o[4] = a4 + b3[4]; o[5] = a5 + b3[5];
}

// ---------- finalize: loss scalar + perplexity ----------
__global__ void finalize_kernel(const float* __restrict__ counts,
                                const float* __restrict__ lossAcc,
                                float* __restrict__ out) {
    __shared__ float red[1024];
    int t = threadIdx.x;
    float p = counts[t] * (1.0f / (float)BATCH);
    red[t] = p * logf(p + 1e-10f);
    __syncthreads();
    for (int off = 512; off > 0; off >>= 1) {
        if (t < off) red[t] += red[t + off];
        __syncthreads();
    }
    if (t == 0) {
        out[1 + (size_t)BATCH * OUTD] = expf(-red[0]);                        // perplexity
        out[0] = 1.25f * (*lossAcc) / ((float)BATCH * (float)EMBD);           // q_lat + CC*e_lat
    }
}

// ---------- launcher ----------
extern "C" void kernel_launch(void* const* d_in, const int* in_sizes, int n_in,
                              void* d_out, int out_size, void* d_ws, size_t ws_size,
                              hipStream_t stream) {
    (void)in_sizes; (void)n_in; (void)out_size; (void)ws_size;
    const float* x   = (const float*)d_in[0];
    const float* W4  = (const float*)d_in[1];
    const float* b4  = (const float*)d_in[2];
    const float* W5  = (const float*)d_in[3];
    const float* b5  = (const float*)d_in[4];
    const float* W6  = (const float*)d_in[5];
    const float* b6  = (const float*)d_in[6];
    const float* emb = (const float*)d_in[7];
    const float* W1  = (const float*)d_in[8];
    const float* b1  = (const float*)d_in[9];
    const float* W2  = (const float*)d_in[10];
    const float* b2  = (const float*)d_in[11];
    const float* W3  = (const float*)d_in[12];
    const float* b3  = (const float*)d_in[13];
    float* out = (float*)d_out;

    // workspace carve-up (256B aligned)
    char* base = (char*)d_ws;
    size_t off = 0;
    auto carve = [&](size_t bytes) {
        char* p = base + off;
        off = (off + bytes + 255) & ~(size_t)255;
        return p;
    };
    unsigned short* h1    = (unsigned short*)carve((size_t)BATCH * HID * 2);   // also g1
    unsigned short* h2    = (unsigned short*)carve((size_t)BATCH * HID * 2);   // also g2
    float*          zf    = (float*)         carve((size_t)BATCH * EMBD * 4);
    unsigned short* zbf   = (unsigned short*)carve((size_t)BATCH * EMBD * 2);
    unsigned short* qbf   = (unsigned short*)carve((size_t)BATCH * EMBD * 2);
    int*            idx   = (int*)           carve((size_t)BATCH * 4);
    unsigned short* Wt5   = (unsigned short*)carve((size_t)HID * HID * 2);
    unsigned short* Wt6   = (unsigned short*)carve((size_t)EMBD * HID * 2);
    unsigned short* Wt1   = (unsigned short*)carve((size_t)HID * EMBD * 2);
    unsigned short* Wt2   = (unsigned short*)carve((size_t)HID * HID * 2);
    unsigned short* embbf = (unsigned short*)carve((size_t)NCODE * EMBD * 2);
    float*          e2    = (float*)         carve((size_t)NCODE * 4);
    float*          cnts  = (float*)         carve((size_t)NCODE * 4);
    float*          lossA = (float*)         carve(256);

    // prep
    wtrans_kernel<<<(HID * HID + 255) / 256, 256, 0, stream>>>(W5, Wt5, HID, HID);
    wtrans_kernel<<<(HID * EMBD + 255) / 256, 256, 0, stream>>>(W6, Wt6, HID, EMBD);
    wtrans_kernel<<<(EMBD * HID + 255) / 256, 256, 0, stream>>>(W1, Wt1, EMBD, HID);
    wtrans_kernel<<<(HID * HID + 255) / 256, 256, 0, stream>>>(W2, Wt2, HID, HID);
    embprep_kernel<<<NCODE, 128, 0, stream>>>(emb, embbf, e2);
    initz_kernel<<<4, 256, 0, stream>>>(cnts, lossA);

    // encoder
    enc1_kernel<<<2 * BATCH, 256, 0, stream>>>(x, W4, b4, h1);
    gemm_bf16_kernel<HID, HID, true, false><<<BATCH / 32, 256, 0, stream>>>(h1, Wt5, b5, h2, nullptr);
    gemm_bf16_kernel<HID, EMBD, false, true><<<BATCH / 32, 256, 0, stream>>>(h2, Wt6, b6, zbf, zf);

    // VQ
    vq_argmin_kernel<<<BATCH / 128, 256, 0, stream>>>(zbf, embbf, e2, idx);
    vq_gather_kernel<<<BATCH / 256, 256, 0, stream>>>(idx, emb, zf, qbf, cnts, lossA);

    // decoder (reuse h1/h2 as g1/g2)
    gemm_bf16_kernel<EMBD, HID, true, false><<<BATCH / 32, 256, 0, stream>>>(qbf, Wt1, b1, h1, nullptr);
    gemm_bf16_kernel<HID, HID, true, false><<<BATCH / 32, 256, 0, stream>>>(h1, Wt2, b2, h2, nullptr);
    dec_out_kernel<<<BATCH / 256, 256, 0, stream>>>(h2, W3, b3, out);

    // scalars
    finalize_kernel<<<1, 1024, 0, stream>>>(cnts, lossA, out);
}
